// MultiHeadCrossAttention_74045236183778
// MI455X (gfx1250) — compile-verified, matrix operand order checked by reading, not run on hardware
//
#include <hip/hip_runtime.h>

#define B_  32
#define S_  1024
#define HN_ 16
#define H_  1024
#define HD_ 64
#define DI_ 768

typedef __attribute__((ext_vector_type(16))) __bf16 v16bf;
typedef __attribute__((ext_vector_type(8)))  float  v8f;
typedef unsigned int u32x4 __attribute__((ext_vector_type(4)));
typedef int          i32x8 __attribute__((ext_vector_type(8)));
typedef int          i32x4 __attribute__((ext_vector_type(4)));

union FragU { v16bf v; unsigned short u[16]; };

__device__ inline v8f v8f_zero() {
  v8f z = {0.f, 0.f, 0.f, 0.f, 0.f, 0.f, 0.f, 0.f};
  return z;
}

__device__ inline unsigned short f32_to_bf16(float x) {
  unsigned int u = __builtin_bit_cast(unsigned int, x);
  u += 0x7FFFu + ((u >> 16) & 1u);           // round-to-nearest-even
  return (unsigned short)(u >> 16);
}
__device__ inline float bf16_to_f32(unsigned short h) {
  unsigned int u = ((unsigned int)h) << 16;
  return __builtin_bit_cast(float, u);
}

__device__ inline v8f wmma_bf16(v16bf a, v16bf b, v8f c) {
  return __builtin_amdgcn_wmma_f32_16x16x32_bf16(
      /*neg_a=*/false, a, /*neg_b=*/false, b,
      /*c_mod=*/(short)0, c, /*reuse_a=*/false, /*reuse_b=*/false);
}

// A-operand fragment (16x32 bf16, MxK) from row-major source, leading dim ld
// (elements). Also used as the B operand when the KxN matrix is available
// transposed (i.e. NxK row-major), which is the case for all our weights.
// ISA layout: lanes 0-15 hold M=lane; lanes 16-31 M=lane-16.
// VGPR v packs K pair k,k+1 with k = (v>>2)*16 + half*8 + (v&3)*2.
__device__ inline v16bf frag_ld_rowmajor(const unsigned short* p, int ld) {
  const int lane = threadIdx.x & 31;
  const int row  = lane & 15;
  const int half = lane >> 4;
  FragU f;
  const unsigned short* r = p + (size_t)row * ld;
#pragma unroll
  for (int v = 0; v < 8; ++v) {
    int k = ((v >> 2) << 4) + (half << 3) + ((v & 3) << 1);
    f.u[2 * v]     = r[k];
    f.u[2 * v + 1] = r[k + 1];
  }
  return f.v;
}

// B-operand fragment from a KxN row-major source (K=32, N=16 tile):
// lane holds column n = lane&15, same K packing as A.
__device__ inline v16bf frag_ld_kxn(const unsigned short* p, int ld) {
  const int lane = threadIdx.x & 31;
  const int col  = lane & 15;
  const int half = lane >> 4;
  FragU f;
#pragma unroll
  for (int v = 0; v < 8; ++v) {
    int k = ((v >> 2) << 4) + (half << 3) + ((v & 3) << 1);
    f.u[2 * v]     = p[(size_t)k * ld + col];
    f.u[2 * v + 1] = p[(size_t)(k + 1) * ld + col];
  }
  return f.v;
}

// ----------------------------------------------------------- TDM staging
// Load a 128x64 bf16 tile (contiguous rows) from global into LDS using the
// Tensor Data Mover.  Descriptor packed per CDNA5 ISA section 8 (D# groups).
// This toolchain exposes the 6-arg builtin:
//   (uint32x4 g0, int32x8 g1, int32x4 g2, int32x4 g3, int32x8 g4, i32 cpol)
__device__ inline void tdm_load_tile_128x64(const unsigned short* src,
                                            unsigned short* lds_dst) {
#if defined(__has_builtin) && __has_builtin(__builtin_amdgcn_tensor_load_to_lds)
  unsigned long long ga = (unsigned long long)(const void*)src;
  unsigned lds = (unsigned)(unsigned long long)(void*)lds_dst;  // LDS byte offset
  u32x4 g0;
  g0[0] = 1u;                                         // count=1, user descriptor
  g0[1] = lds;                                        // lds_addr
  g0[2] = (unsigned)(ga & 0xFFFFFFFFu);               // global_addr[31:0]
  g0[3] = (unsigned)((ga >> 32) & 0x01FFFFFFu) | (2u << 30);  // addr[56:32]|type=2
  i32x8 g1;
  g1[0] = 0x00010000;          // workgroup_mask=0, data_size=1 (2 bytes)
  g1[1] = (int)(64u << 16);    // tensor_dim0 = 64 (bits 79:48, low half)
  g1[2] = (int)(128u << 16);   // tensor_dim1 = 128 (bits 111:80, low half)
  g1[3] = (int)(64u << 16);    // tile_dim0 = 64 (bits 127:112)
  g1[4] = 128;                 // tile_dim1 = 128 (bits 143:128)
  g1[5] = 64;                  // tensor_dim0_stride = 64 elems (bits 207:160)
  g1[6] = 0;
  g1[7] = 0;
  i32x4 gz;
  gz[0] = 0; gz[1] = 0; gz[2] = 0; gz[3] = 0;
  i32x8 gz8;
  gz8[0] = 0; gz8[1] = 0; gz8[2] = 0; gz8[3] = 0;
  gz8[4] = 0; gz8[5] = 0; gz8[6] = 0; gz8[7] = 0;
  __builtin_amdgcn_tensor_load_to_lds(g0, g1, gz, gz, gz8, 0);
#else
  for (int i = threadIdx.x & 31; i < 128 * 64; i += 32) lds_dst[i] = src[i];
#endif
}

__device__ inline void tdm_wait() {
#if defined(__has_builtin) && __has_builtin(__builtin_amdgcn_s_wait_tensorcnt)
  __builtin_amdgcn_s_wait_tensorcnt(0);
#endif
}

// ---------------------------------------------------------------- converts
__global__ void cvt_bf16_kernel(const float* __restrict__ src,
                                unsigned short* __restrict__ dst, int n) {
  int i = blockIdx.x * blockDim.x + threadIdx.x;
  int stride = gridDim.x * blockDim.x;
  for (; i < n; i += stride) dst[i] = f32_to_bf16(src[i]);
}

// ------------------------------------------------- projection GEMM (bf16 WMMA)
// C[M=32768, N=1024] = X[M,768] * W[N,768]^T + bias[N]
// Block = 4 waves; block tile 128(M) x 64(N); wave tile 32(M) x 64(N).
// Software-pipelined: fragments for k+1 load while k's WMMAs execute.
__global__ void __launch_bounds__(128)
gemm_proj_kernel(const unsigned short* __restrict__ X,
                 const unsigned short* __restrict__ W,
                 const float* __restrict__ bias,
                 float* __restrict__ out) {
  const int wave = threadIdx.x >> 5;
  const int lane = threadIdx.x & 31;
  const int m0 = blockIdx.x * 128 + wave * 32;
  const int n0 = blockIdx.y * 64;

  const unsigned short* Xp = X + (size_t)m0 * DI_;
  const unsigned short* Wp = W + (size_t)n0 * DI_;

  v8f acc[2][4];
#pragma unroll
  for (int mt = 0; mt < 2; ++mt)
#pragma unroll
    for (int nt = 0; nt < 4; ++nt) acc[mt][nt] = v8f_zero();

  // prologue: fragments for k = 0
  v16bf a[2], b[4];
#pragma unroll
  for (int mt = 0; mt < 2; ++mt)
    a[mt] = frag_ld_rowmajor(Xp + (size_t)mt * 16 * DI_, DI_);
#pragma unroll
  for (int nt = 0; nt < 4; ++nt)
    b[nt] = frag_ld_rowmajor(Wp + (size_t)nt * 16 * DI_, DI_);

  for (int kk = 0; kk < DI_; kk += 32) {
    // issue next iteration's loads (wrap on last iter; harmless redundant load)
    const int kn = (kk + 32 < DI_) ? (kk + 32) : 0;
    __builtin_prefetch(Xp + kk + 96, 0, 1);             // speculative prefetch
    v16bf an[2], bn[4];
#pragma unroll
    for (int mt = 0; mt < 2; ++mt)
      an[mt] = frag_ld_rowmajor(Xp + (size_t)mt * 16 * DI_ + kn, DI_);
#pragma unroll
    for (int nt = 0; nt < 4; ++nt)
      bn[nt] = frag_ld_rowmajor(Wp + (size_t)nt * 16 * DI_ + kn, DI_);

    // compute with current fragments while next loads are in flight
#pragma unroll
    for (int mt = 0; mt < 2; ++mt)
#pragma unroll
      for (int nt = 0; nt < 4; ++nt)
        acc[mt][nt] = wmma_bf16(a[mt], b[nt], acc[mt][nt]);

#pragma unroll
    for (int mt = 0; mt < 2; ++mt) a[mt] = an[mt];
#pragma unroll
    for (int nt = 0; nt < 4; ++nt) b[nt] = bn[nt];
  }

  const int nlo = lane & 15;
  const int half = lane >> 4;
#pragma unroll
  for (int mt = 0; mt < 2; ++mt) {
    const int rbase = m0 + mt * 16 + half * 8;          // C layout: M = r + 8*half
#pragma unroll
    for (int nt = 0; nt < 4; ++nt) {
      int n = n0 + nt * 16 + nlo;
      float bb = bias[n];
#pragma unroll
      for (int r = 0; r < 8; ++r)
        out[(size_t)(rbase + r) * H_ + n] = acc[mt][nt][r] + bb;
    }
  }
}

// ------------------------------------- LayerNorm + L2 norm + scale + relayout
__device__ inline float block_sum_256(float v) {
  __shared__ float red[8];
#pragma unroll
  for (int m = 16; m >= 1; m >>= 1) v += __shfl_xor(v, m, 32);
  __syncthreads();
  if ((threadIdx.x & 31) == 0) red[threadIdx.x >> 5] = v;
  __syncthreads();
  float t = 0.f;
#pragma unroll
  for (int i = 0; i < 8; ++i) t += red[i];
  return t;
}

__global__ void __launch_bounds__(256)
ln_l2_relayout_kernel(const float* __restrict__ X,        // [B*S, 1024]
                      const float* __restrict__ gamma,
                      const float* __restrict__ beta,
                      unsigned short* __restrict__ out_bf) { // [h][s][b][64]
  const int row = blockIdx.x;     // row = b*S + s
  const int b = row >> 10;
  const int s = row & (S_ - 1);
  const int t = threadIdx.x;

  float x[4];
  float sum = 0.f;
#pragma unroll
  for (int i = 0; i < 4; ++i) {
    x[i] = X[(size_t)row * H_ + t * 4 + i];
    sum += x[i];
  }
  float mu = block_sum_256(sum) * (1.0f / H_);

  float sq = 0.f;
#pragma unroll
  for (int i = 0; i < 4; ++i) { float d = x[i] - mu; sq += d * d; }
  float var = block_sum_256(sq) * (1.0f / H_);
  float rstd = rsqrtf(var + 1e-5f);

  float y[4];
  float l2 = 0.f;
#pragma unroll
  for (int i = 0; i < 4; ++i) {
    int j = t * 4 + i;
    y[i] = (x[i] - mu) * rstd * gamma[j] + beta[j];
    l2 += y[i] * y[i];
  }
  float nrm = sqrtf(block_sum_256(l2));
  float inv = 0.125f / fmaxf(nrm, 1e-12f);   // L2-normalize then HD^-0.5 scale

#pragma unroll
  for (int i = 0; i < 4; ++i) {
    int j = t * 4 + i;
    int h = j >> 6, e = j & 63;
    out_bf[(((size_t)h * S_ + s) * B_ + b) * HD_ + e] = f32_to_bf16(y[i] * inv);
  }
}

// --------------------------------------------------- fused batch-attention
// One wave handles one (h, s): QKV proj, 32x32 softmax, ctx, out-proj+residual.
// X tiles for the whole block are staged into LDS with the Tensor Data Mover.
#define SWAVES 4

// X (32x64 bf16 in LDS, ld 64) times W^T (W is 64x64 NxK bf16) + bias -> dst.
// May operate in place (dst == X): per-mt reads complete before that mt's
// writes are issued, and rows of other mts are untouched until their turn.
__device__ inline void proj_32x64(const unsigned short* X,
                                  const unsigned short* W,
                                  const float* bias,
                                  unsigned short* dst) {
  const int lane = threadIdx.x & 31;
  // hoist all 8 weight fragments (64x64 NxK = 4 N-tiles x 2 K-steps)
  v16bf b[8];
#pragma unroll
  for (int nt = 0; nt < 4; ++nt) {
    b[2 * nt]     = frag_ld_rowmajor(W + nt * 16 * 64, 64);
    b[2 * nt + 1] = frag_ld_rowmajor(W + nt * 16 * 64 + 32, 64);
  }
#pragma unroll
  for (int mt = 0; mt < 2; ++mt) {
    v16bf a0 = frag_ld_rowmajor(X + mt * 16 * 64, 64);
    v16bf a1 = frag_ld_rowmajor(X + mt * 16 * 64 + 32, 64);
#pragma unroll
    for (int nt = 0; nt < 4; ++nt) {
      v8f acc = v8f_zero();
      acc = wmma_bf16(a0, b[2 * nt], acc);
      acc = wmma_bf16(a1, b[2 * nt + 1], acc);
      int n = nt * 16 + (lane & 15);
      float bb = bias[n];
      int mrow = mt * 16 + (lane >> 4) * 8;
#pragma unroll
      for (int r = 0; r < 8; ++r)
        dst[(mrow + r) * 64 + n] = f32_to_bf16(acc[r] + bb);
    }
  }
}

__global__ void __launch_bounds__(128)
cross_attn_kernel(const unsigned short* __restrict__ Xq,   // [16][1024][32][64]
                  const unsigned short* __restrict__ Xkv,
                  const unsigned short* __restrict__ Win,  // [16][192][64]
                  const float* __restrict__ Bin,           // [16][192]
                  const unsigned short* __restrict__ Wout, // [16][64][64]
                  const float* __restrict__ Bout,          // [16][64]
                  float* __restrict__ out,                 // (B,Hn,S,128)
                  int side) {
  __shared__ unsigned short shQ[SWAVES][32 * 64];  // staged Xq -> Q -> ctx
  __shared__ unsigned short shK[SWAVES][32 * 64];  // staged Xkv -> K
  __shared__ unsigned short shV[SWAVES][32 * 64];  // V
  __shared__ unsigned short shA[SWAVES][32 * 32];  // attn weights

  const int wave = threadIdx.x >> 5;
  const int lane = threadIdx.x & 31;
  const int h = blockIdx.y;
  const int s0 = blockIdx.x * SWAVES;              // first s of this block
  const int s = s0 + wave;

  const unsigned short* xq  = Xq  + (((size_t)h * S_ + s) * B_) * HD_;
  const unsigned short* wq = Win + (size_t)h * 192 * HD_;
  const unsigned short* wk = wq + 64 * 64;
  const unsigned short* wv = wq + 2 * 64 * 64;
  const float* bq = Bin + h * 192;

  // ---- TDM: stage this block's 4 consecutive (h,s) tiles (128x64 each side)
  if (wave == 0) {
    tdm_load_tile_128x64(Xq  + (((size_t)h * S_ + s0) * B_) * HD_, &shQ[0][0]);
    tdm_load_tile_128x64(Xkv + (((size_t)h * S_ + s0) * B_) * HD_, &shK[0][0]);
    tdm_wait();
  }
  __syncthreads();

  // ---- QKV projections from staged tiles (48 WMMAs).  Order matters:
  // V first (reads staged Xkv in shK), then K in place, then Q in place.
  proj_32x64(shK[wave], wv, bq + 128, shV[wave]);
  proj_32x64(shK[wave], wk, bq + 64,  shK[wave]);
  proj_32x64(shQ[wave], wq, bq,       shQ[wave]);

  // ---- scores = (Q*0.125) K^T : 32x32 over K=64 (8 WMMAs)
  v8f sc[2][2];
  {
    v16bf kb[4];
#pragma unroll
    for (int nt = 0; nt < 2; ++nt) {
      kb[2 * nt]     = frag_ld_rowmajor(shK[wave] + nt * 16 * 64, 64);
      kb[2 * nt + 1] = frag_ld_rowmajor(shK[wave] + nt * 16 * 64 + 32, 64);
    }
#pragma unroll
    for (int mt = 0; mt < 2; ++mt) {
      v16bf a0 = frag_ld_rowmajor(shQ[wave] + mt * 16 * 64, 64);
      v16bf a1 = frag_ld_rowmajor(shQ[wave] + mt * 16 * 64 + 32, 64);
#pragma unroll
      for (int nt = 0; nt < 2; ++nt) {
        v8f acc = v8f_zero();
        acc = wmma_bf16(a0, kb[2 * nt], acc);
        acc = wmma_bf16(a1, kb[2 * nt + 1], acc);
        sc[mt][nt] = acc;
      }
    }
  }

  // ---- softmax over 32 columns; row of a 16x16 C tile lives on 16 lanes
  const int half = lane >> 4;
  const int nlo = lane & 15;
#pragma unroll
  for (int mt = 0; mt < 2; ++mt) {
#pragma unroll
    for (int r = 0; r < 8; ++r) {
      float v0 = sc[mt][0][r] * 0.125f;
      float v1 = sc[mt][1][r] * 0.125f;
      float mx = fmaxf(v0, v1);
#pragma unroll
      for (int msk = 8; msk >= 1; msk >>= 1) mx = fmaxf(mx, __shfl_xor(mx, msk, 32));
      float e0 = __expf(v0 - mx);
      float e1 = __expf(v1 - mx);
      float sm = e0 + e1;
#pragma unroll
      for (int msk = 8; msk >= 1; msk >>= 1) sm += __shfl_xor(sm, msk, 32);
      float inv = 1.0f / sm;
      int m = mt * 16 + half * 8 + r;
      shA[wave][m * 32 + nlo]      = f32_to_bf16(e0 * inv);
      shA[wave][m * 32 + 16 + nlo] = f32_to_bf16(e1 * inv);
    }
  }

  // ---- ctx = attn(32x32) * V(32x64)  (8 WMMAs), write ctx into shQ
  v8f cx[2][4];
  {
    v16bf vb[4];
#pragma unroll
    for (int nt = 0; nt < 4; ++nt)
      vb[nt] = frag_ld_kxn(shV[wave] + nt * 16, 64);         // V is KxN row-major
#pragma unroll
    for (int mt = 0; mt < 2; ++mt) {
      v16bf a = frag_ld_rowmajor(shA[wave] + mt * 16 * 32, 32);  // K=32 in one frag
#pragma unroll
      for (int nt = 0; nt < 4; ++nt) {
        v8f acc = v8f_zero();
        cx[mt][nt] = wmma_bf16(a, vb[nt], acc);
      }
    }
  }
#pragma unroll
  for (int mt = 0; mt < 2; ++mt)
#pragma unroll
    for (int nt = 0; nt < 4; ++nt) {
      int n = nt * 16 + nlo;
      int mrow = mt * 16 + half * 8;
#pragma unroll
      for (int r = 0; r < 8; ++r)
        shQ[wave][(mrow + r) * 64 + n] = f32_to_bf16(cx[mt][nt][r]);
    }

  // ---- out-proj + bias + residual (16 WMMAs)
  const unsigned short* wo = Wout + (size_t)h * 64 * 64;
  const float* bo = Bout + h * 64;
  {
    v16bf ob[8];
#pragma unroll
    for (int nt = 0; nt < 4; ++nt) {
      ob[2 * nt]     = frag_ld_rowmajor(wo + nt * 16 * 64, 64);
      ob[2 * nt + 1] = frag_ld_rowmajor(wo + nt * 16 * 64 + 32, 64);
    }
#pragma unroll
    for (int mt = 0; mt < 2; ++mt) {
      v16bf a0 = frag_ld_rowmajor(shQ[wave] + mt * 16 * 64, 64);
      v16bf a1 = frag_ld_rowmajor(shQ[wave] + mt * 16 * 64 + 32, 64);
#pragma unroll
      for (int nt = 0; nt < 4; ++nt) {
        v8f acc = v8f_zero();
        acc = wmma_bf16(a0, ob[2 * nt], acc);
        acc = wmma_bf16(a1, ob[2 * nt + 1], acc);
        int n = nt * 16 + nlo;
        float bb = bo[n];
        int mrow = mt * 16 + half * 8;
#pragma unroll
        for (int r = 0; r < 8; ++r) {
          int bidx = mrow + r;  // batch index
          float res = bf16_to_f32(xq[bidx * 64 + n]);
          out[(((size_t)bidx * HN_ + h) * S_ + s) * 128 + side + n] = acc[r] + bb + res;
        }
      }
    }
  }
}

// ------------------------------------------------ final LN over concat(128)
__global__ void __launch_bounds__(256)
final_ln_kernel(float* __restrict__ out,
                const float* __restrict__ g, const float* __restrict__ b) {
  const int wave = threadIdx.x >> 5;
  const int lane = threadIdx.x & 31;
  const size_t row = (size_t)blockIdx.x * 8 + wave;   // (bb*Hn + h)*S + s
  const int h = (int)((row >> 10) & (HN_ - 1));
  float* p = out + row * 128;

  float x[4];
  float sum = 0.f;
#pragma unroll
  for (int i = 0; i < 4; ++i) { x[i] = p[lane * 4 + i]; sum += x[i]; }
#pragma unroll
  for (int m = 16; m >= 1; m >>= 1) sum += __shfl_xor(sum, m, 32);
  float mu = sum * (1.0f / 128.0f);

  float sq = 0.f;
#pragma unroll
  for (int i = 0; i < 4; ++i) { float d = x[i] - mu; sq += d * d; }
#pragma unroll
  for (int m = 16; m >= 1; m >>= 1) sq += __shfl_xor(sq, m, 32);
  float rstd = rsqrtf(sq * (1.0f / 128.0f) + 1e-5f);

#pragma unroll
  for (int i = 0; i < 4; ++i) {
    int j = lane * 4 + i;
    p[j] = (x[i] - mu) * rstd * g[h * 128 + j] + b[h * 128 + j];
  }
}

// ---------------------------------------------------------------- launcher
extern "C" void kernel_launch(void* const* d_in, const int* in_sizes, int n_in,
                              void* d_out, int out_size, void* d_ws, size_t ws_size,
                              hipStream_t stream) {
  (void)in_sizes; (void)n_in; (void)out_size; (void)ws_size;

  char* base = (char*)d_ws;
  size_t off = 0;
  auto take = [&](size_t bytes) -> void* {
    void* p = base + off;
    off += (bytes + 255) & ~(size_t)255;
    return p;
  };

  unsigned short* imgF_bf   = (unsigned short*)take((size_t)B_ * S_ * DI_ * 2);
  unsigned short* txtF_bf   = (unsigned short*)take((size_t)B_ * S_ * DI_ * 2);
  unsigned short* wimg_bf   = (unsigned short*)take((size_t)H_ * DI_ * 2);
  unsigned short* wtxt_bf   = (unsigned short*)take((size_t)H_ * DI_ * 2);
  unsigned short* i2t_iw_bf = (unsigned short*)take((size_t)HN_ * 192 * HD_ * 2);
  unsigned short* i2t_ow_bf = (unsigned short*)take((size_t)HN_ * HD_ * HD_ * 2);
  unsigned short* t2i_iw_bf = (unsigned short*)take((size_t)HN_ * 192 * HD_ * 2);
  unsigned short* t2i_ow_bf = (unsigned short*)take((size_t)HN_ * HD_ * HD_ * 2);
  float*          proj_tmp  = (float*)take((size_t)B_ * S_ * H_ * 4);
  unsigned short* imgh_bf   = (unsigned short*)take((size_t)HN_ * S_ * B_ * HD_ * 2);
  unsigned short* txth_bf   = (unsigned short*)take((size_t)HN_ * S_ * B_ * HD_ * 2);

  auto cvt = [&](int idx, unsigned short* dst, int n) {
    int blocks = (n + 1023) / 1024;
    if (blocks > 16384) blocks = 16384;
    cvt_bf16_kernel<<<blocks, 256, 0, stream>>>((const float*)d_in[idx], dst, n);
  };

  cvt(0,  imgF_bf,   B_ * S_ * DI_);
  cvt(1,  txtF_bf,   B_ * S_ * DI_);
  cvt(2,  wimg_bf,   H_ * DI_);
  cvt(6,  wtxt_bf,   H_ * DI_);
  cvt(10, i2t_iw_bf, HN_ * 192 * HD_);
  cvt(12, i2t_ow_bf, HN_ * HD_ * HD_);
  cvt(14, t2i_iw_bf, HN_ * 192 * HD_);
  cvt(16, t2i_ow_bf, HN_ * HD_ * HD_);

  dim3 gGemm(B_ * S_ / 128, H_ / 64);  // (256, 16)
  dim3 gAttn(S_ / SWAVES, HN_);        // (256, 16)

  // image branch: proj -> LN/L2/relayout
  gemm_proj_kernel<<<gGemm, 128, 0, stream>>>(imgF_bf, wimg_bf, (const float*)d_in[3], proj_tmp);
  ln_l2_relayout_kernel<<<B_ * S_, 256, 0, stream>>>(proj_tmp, (const float*)d_in[4],
                                                     (const float*)d_in[5], imgh_bf);
  // text branch
  gemm_proj_kernel<<<gGemm, 128, 0, stream>>>(txtF_bf, wtxt_bf, (const float*)d_in[7], proj_tmp);
  ln_l2_relayout_kernel<<<B_ * S_, 256, 0, stream>>>(proj_tmp, (const float*)d_in[8],
                                                     (const float*)d_in[9], txth_bf);

  // fused cross attention: i2t into cols [0,64), t2i into cols [64,128)
  cross_attn_kernel<<<gAttn, 128, 0, stream>>>(imgh_bf, txth_bf, i2t_iw_bf,
                                               (const float*)d_in[11], i2t_ow_bf,
                                               (const float*)d_in[13], (float*)d_out, 0);
  cross_attn_kernel<<<gAttn, 128, 0, stream>>>(txth_bf, imgh_bf, t2i_iw_bf,
                                               (const float*)d_in[15], t2i_ow_bf,
                                               (const float*)d_in[17], (float*)d_out, 64);

  // final LayerNorm over the concatenated 128-dim, in place on d_out
  final_ln_kernel<<<(B_ * HN_ * S_) / 8, 256, 0, stream>>>((float*)d_out,
                                                           (const float*)d_in[18],
                                                           (const float*)d_in[19]);
}
